// GPT_67018669686791
// MI455X (gfx1250) — compile-verified, matrix operand order checked by reading, not run on hardware
//
#include <hip/hip_runtime.h>
#include <hip/hip_bf16.h>
#include <math.h>
#include <stdint.h>

typedef __attribute__((ext_vector_type(16))) _Float16 v16h;
typedef __attribute__((ext_vector_type(8)))  _Float16 h8v;
typedef __attribute__((ext_vector_type(8)))  float    v8f;

#define NL   12
#define NH   16
#define Dm   1024
#define DFF  4096
#define B_   2
#define S_   1024
#define DH   64

// ---------------------------------------------------------------------------
// WMMA helper: D = A(16x32 f16) * B(32x16 f16) + C(16x16 f32)
// ---------------------------------------------------------------------------
__device__ __forceinline__ v8f wmma16(v16h a, v16h b, v8f c) {
  return __builtin_amdgcn_wmma_f32_16x16x32_f16(false, a, false, b, (short)0, c,
                                                false, false);
}

// Load a 16-bit A/B fragment from a row-major f16 tile (LDS or global).
// CDNA5 16-bit fragment layout: lane L holds row/col (sel + L%16);
// halves 0..7 -> K = g+0..7, halves 8..15 -> K = g+16..23, g = k0 + 8*(L>=16).
// All call sites keep p 16-byte aligned, so this is two b128 loads.
__device__ __forceinline__ v16h ld_frag16(const _Float16* p0, int pitch,
                                          int sel, int k0) {
  const int lane = threadIdx.x & 31;
  const _Float16* p =
      p0 + (size_t)(sel + (lane & 15)) * pitch + k0 + ((lane & 16) ? 8 : 0);
  h8v lo = *(const h8v*)p;
  h8v hi = *(const h8v*)(p + 16);
  v16h f;
#pragma unroll
  for (int i = 0; i < 8; ++i) {
    f[i] = lo[i];
    f[i + 8] = hi[i];
  }
  return f;
}

__device__ __forceinline__ float gelu_f(float x) {
  const float c = 0.7978845608028654f;  // sqrt(2/pi)
  return 0.5f * x * (1.0f + tanhf(c * (x + 0.044715f * x * x * x)));
}

// ---------------------------------------------------------------------------
// Embedding: h[b,s,:] = wte[id] + wpe[s]   (f32 residual stream)
// ---------------------------------------------------------------------------
__global__ __launch_bounds__(256) void embed_kernel(
    const int* __restrict__ ids, const float* __restrict__ wte,
    const float* __restrict__ wpe, float* __restrict__ h) {
  const size_t idx = (size_t)blockIdx.x * 256 + threadIdx.x;
  const size_t tok = idx / Dm;
  const int d = (int)(idx % Dm);
  const int s = (int)(tok % S_);
  const int id = ids[tok];
  h[idx] = wte[(size_t)id * Dm + d] + wpe[(size_t)s * Dm + d];
}

// ---------------------------------------------------------------------------
// Weight convert + transpose: src f32 [K][N] -> dst f16 [N][K]
// Grid: (N/64, K/64), 256 threads, 64x64 tile through LDS.
// ---------------------------------------------------------------------------
__global__ __launch_bounds__(256) void wcvt_kernel(const float* __restrict__ src,
                                                   _Float16* __restrict__ dst,
                                                   int K, int N) {
  __shared__ _Float16 t[64][72];  // [n][k], 144B row pitch (16B aligned)
  const int tid = threadIdx.x;
  const int n0 = blockIdx.x * 64, k0 = blockIdx.y * 64;
#pragma unroll
  for (int it = 0; it < 4; ++it) {
    int i = tid + it * 256;  // 1024 slots: 64 k-rows x 16 col4s
    int k = i >> 4, n4 = (i & 15) << 2;
    const float4 v = *(const float4*)(src + (size_t)(k0 + k) * N + n0 + n4);
    t[n4 + 0][k] = (_Float16)v.x;
    t[n4 + 1][k] = (_Float16)v.y;
    t[n4 + 2][k] = (_Float16)v.z;
    t[n4 + 3][k] = (_Float16)v.w;
  }
  __syncthreads();
#pragma unroll
  for (int it = 0; it < 2; ++it) {
    int i = tid + it * 256;  // 512 slots: 64 n-rows x 8 chunk8s
    int n = i >> 3, c8 = (i & 7) << 3;
    *(h8v*)(dst + (size_t)(n0 + n) * K + k0 + c8) = *(const h8v*)&t[n][c8];
  }
}

// ---------------------------------------------------------------------------
// LayerNorm over D=1024, one block (256 threads) per row; templated output
// ---------------------------------------------------------------------------
template <typename OT>
__global__ __launch_bounds__(256) void ln_kernel(const float* __restrict__ x,
                                                 const float* __restrict__ g,
                                                 const float* __restrict__ bta,
                                                 OT* __restrict__ y) {
  __shared__ float red[256];
  const int tid = threadIdx.x;
  const size_t row = blockIdx.x;
  const float* xr = x + row * Dm;
  float v[4];
  float s = 0.0f;
#pragma unroll
  for (int i = 0; i < 4; ++i) {
    v[i] = xr[tid + i * 256];
    s += v[i];
  }
  red[tid] = s;
  __syncthreads();
  for (int st = 128; st > 0; st >>= 1) {
    if (tid < st) red[tid] += red[tid + st];
    __syncthreads();
  }
  const float mean = red[0] * (1.0f / Dm);
  __syncthreads();
  float sq = 0.0f;
#pragma unroll
  for (int i = 0; i < 4; ++i) {
    float d = v[i] - mean;
    sq += d * d;
  }
  red[tid] = sq;
  __syncthreads();
  for (int st = 128; st > 0; st >>= 1) {
    if (tid < st) red[tid] += red[tid + st];
    __syncthreads();
  }
  const float rstd = rsqrtf(red[0] * (1.0f / Dm) + 1e-5f);
  OT* yr = y + row * Dm;
#pragma unroll
  for (int i = 0; i < 4; ++i) {
    int d = tid + i * 256;
    yr[d] = (OT)(g[d] * (v[i] - mean) * rstd + bta[d]);
  }
}

// ---------------------------------------------------------------------------
// WMMA GEMM: C = act(A[M,K]f16 @ Wt[N,K]f16^T + bias) (+ f32 resid)
// Block 256 thr = 8 waves; tile 128(M)x128(N); K-chunk 64 (2 wmma k-steps).
// A tile staged with GLOBAL_LOAD_ASYNC_TO_LDS_B128 (no VGPR round trip),
// double-buffered with a counted s_wait_asynccnt (async loads complete in
// order: <=4 outstanding per wave => current tile landed, next still flying).
// W fragments read directly from global (pre-transposed f16, per-lane
// contiguous; 192MB L2 absorbs the 16x tile reuse).
// ---------------------------------------------------------------------------
template <bool GELU, bool RESID, bool OUTF16>
__global__ __launch_bounds__(256) void gemm16_kernel(
    const _Float16* __restrict__ A, const _Float16* __restrict__ Wt,
    const float* __restrict__ bias, const float* __restrict__ resid,
    void* __restrict__ Cout, int M, int N, int K) {
  __shared__ _Float16 Al[2][128][72];  // double-buffered [m][k], 144B pitch
  const int tid = threadIdx.x;
  const int lane = tid & 31, w = tid >> 5;
  const int grp = (lane >> 4) & 1;
  const int wm = w & 3, wn = w >> 2;  // 4(M) x 2(N) wave grid
  const int m0 = blockIdx.y * 128, n0 = blockIdx.x * 128;
  const _Float16* wbase = Wt + (size_t)(n0 + wn * 64) * K;

  // issue one 128x64-half tile as 4 per-wave async b128 copies
  auto issue_tile = [&](int kc, int buf) {
#pragma unroll
    for (int it = 0; it < 4; ++it) {
      int i = tid + it * 256;  // 1024 slots: 128 rows x 8 chunk8s
      int r = i >> 3, c8 = (i & 7) << 3;
      // low 32 bits of the flat address == LDS byte address (ISA 10.2)
      uint32_t la = (uint32_t)(uintptr_t)&Al[buf][r][c8];
      const _Float16* gp = A + (size_t)(m0 + r) * K + kc + c8;
      asm volatile("global_load_async_to_lds_b128 %0, %1, off"
                   :
                   : "v"(la), "v"(gp)
                   : "memory");
    }
  };

  v8f acc[2][4];
#pragma unroll
  for (int i = 0; i < 2; ++i)
#pragma unroll
    for (int j = 0; j < 4; ++j)
#pragma unroll
      for (int r = 0; r < 8; ++r) acc[i][j][r] = 0.0f;

  issue_tile(0, 0);
  int buf = 0;
  for (int kc = 0; kc < K; kc += 64) {
    const bool more = (kc + 64 < K);
    if (more) {
      issue_tile(kc + 64, buf ^ 1);
      asm volatile("s_wait_asynccnt 0x4" ::: "memory");
    } else {
      asm volatile("s_wait_asynccnt 0x0" ::: "memory");
    }
    __syncthreads();
#pragma unroll
    for (int k2 = 0; k2 < 2; ++k2) {
      v16h af0 = ld_frag16(&Al[buf][0][0], 72, wm * 32, k2 * 32);
      v16h af1 = ld_frag16(&Al[buf][0][0], 72, wm * 32 + 16, k2 * 32);
#pragma unroll
      for (int ns = 0; ns < 4; ++ns) {
        v16h bf = ld_frag16(wbase + kc + k2 * 32, K, ns * 16, 0);
        acc[0][ns] = wmma16(af0, bf, acc[0][ns]);
        acc[1][ns] = wmma16(af1, bf, acc[1][ns]);
      }
    }
    __syncthreads();
    buf ^= 1;
  }

  // Epilogue: C/D layout -> row = r + 8*grp, col = lane%16
#pragma unroll
  for (int ms = 0; ms < 2; ++ms)
#pragma unroll
    for (int ns = 0; ns < 4; ++ns) {
      int col = n0 + wn * 64 + ns * 16 + (lane & 15);
      float bv = bias[col];
#pragma unroll
      for (int r = 0; r < 8; ++r) {
        int row = m0 + wm * 32 + ms * 16 + r + 8 * grp;
        float v = acc[ms][ns][r] + bv;
        if (GELU) v = gelu_f(v);
        if (RESID) v += resid[(size_t)row * N + col];
        if (OUTF16)
          ((_Float16*)Cout)[(size_t)row * N + col] = (_Float16)v;
        else
          ((float*)Cout)[(size_t)row * N + col] = v;
      }
    }
}

// ---------------------------------------------------------------------------
// Fused causal attention (flash-style, online softmax), f16 in/out.
// Grid: (S/64, B*NH); block 128 thr = 4 waves; wave w owns q-rows w*16..+15,
// so softmax stats are wave-private (vgpr idx r <-> row r+8*laneGrp).
// Q/K fragments come straight from global (per-lane contiguous f16);
// only V (transposed) and P bounce through LDS.
// ---------------------------------------------------------------------------
__global__ __launch_bounds__(128) void attn_kernel(
    const _Float16* __restrict__ qkv, _Float16* __restrict__ aout) {
  __shared__ _Float16 Vt[DH][64 + 8];  // [d][key]
  __shared__ _Float16 Ps[64][64 + 8];  // [q_row][key]
  const int tid = threadIdx.x;
  const int lane = tid & 31, w = tid >> 5;
  const int grp = (lane >> 4) & 1;
  const int q0 = blockIdx.x * 64;
  const int b = blockIdx.y / NH, hh = blockIdx.y % NH;
  const int pitch = 3 * Dm;
  const _Float16* qb = qkv + (size_t)(b * S_ + q0) * pitch + hh * DH;
  const _Float16* kb0 = qkv + (size_t)b * S_ * pitch + Dm + hh * DH;
  const _Float16* vb0 = qkv + (size_t)b * S_ * pitch + 2 * Dm + hh * DH;

  const v16h qa0 = ld_frag16(qb, pitch, w * 16, 0);
  const v16h qa1 = ld_frag16(qb, pitch, w * 16, 32);

  float m_run[8], l_run[8];
  v8f o[4];
#pragma unroll
  for (int r = 0; r < 8; ++r) {
    m_run[r] = -3.0e38f;
    l_run[r] = 0.0f;
  }
#pragma unroll
  for (int ns = 0; ns < 4; ++ns)
#pragma unroll
    for (int r = 0; r < 8; ++r) o[ns][r] = 0.0f;

  const int nkt = blockIdx.x + 1;  // causal: skip fully-masked key tiles
  for (int kt = 0; kt < nkt; ++kt) {
    const int k0 = kt * 64;
    __syncthreads();
    // stage V transposed: Vt[d][key]
    for (int i = tid; i < 64 * (DH / 8); i += 128) {
      int r = i >> 3, c8 = (i & 7) << 3;
      h8v v = *(const h8v*)(vb0 + (size_t)(k0 + r) * pitch + c8);
#pragma unroll
      for (int j = 0; j < 8; ++j) Vt[c8 + j][r] = v[j];
    }
    __syncthreads();

    // S = Q K^T for this wave's 16-row strip (K frags direct from global)
    const _Float16* kb = kb0 + (size_t)k0 * pitch;
    v8f sf[4];
#pragma unroll
    for (int ns = 0; ns < 4; ++ns)
#pragma unroll
      for (int r = 0; r < 8; ++r) sf[ns][r] = 0.0f;
#pragma unroll
    for (int ns = 0; ns < 4; ++ns) {
      sf[ns] = wmma16(qa0, ld_frag16(kb, pitch, ns * 16, 0), sf[ns]);
      sf[ns] = wmma16(qa1, ld_frag16(kb, pitch, ns * 16, 32), sf[ns]);
    }
#pragma unroll
    for (int ns = 0; ns < 4; ++ns)
#pragma unroll
      for (int r = 0; r < 8; ++r) sf[ns][r] *= 0.125f;  // 1/sqrt(dh)
    if (kt == blockIdx.x) {  // diagonal tile: causal mask
#pragma unroll
      for (int ns = 0; ns < 4; ++ns) {
        int col = ns * 16 + (lane & 15);
#pragma unroll
        for (int r = 0; r < 8; ++r) {
          int row = w * 16 + r + 8 * grp;
          if (col > row) sf[ns][r] = -1.0e9f;
        }
      }
    }

    // online softmax (per-row = per vgpr index, per 16-lane group)
    float mnew[8], alpha[8], tsum[8];
#pragma unroll
    for (int r = 0; r < 8; ++r) {
      float t = sf[0][r];
      t = fmaxf(t, sf[1][r]);
      t = fmaxf(t, sf[2][r]);
      t = fmaxf(t, sf[3][r]);
#pragma unroll
      for (int off = 8; off > 0; off >>= 1)
        t = fmaxf(t, __shfl_xor(t, off, 32));
      mnew[r] = fmaxf(m_run[r], t);
      alpha[r] = __expf(m_run[r] - mnew[r]);
      tsum[r] = 0.0f;
    }
#pragma unroll
    for (int ns = 0; ns < 4; ++ns) {
      int col = ns * 16 + (lane & 15);
#pragma unroll
      for (int r = 0; r < 8; ++r) {
        float p = __expf(sf[ns][r] - mnew[r]);
        tsum[r] += p;
        Ps[w * 16 + r + 8 * grp][col] = (_Float16)p;  // wave-private rows
      }
    }
#pragma unroll
    for (int r = 0; r < 8; ++r) {
#pragma unroll
      for (int off = 8; off > 0; off >>= 1)
        tsum[r] += __shfl_xor(tsum[r], off, 32);
      l_run[r] = l_run[r] * alpha[r] + tsum[r];
      m_run[r] = mnew[r];
#pragma unroll
      for (int ns = 0; ns < 4; ++ns) o[ns][r] *= alpha[r];
    }

    // O += P V
    v16h pa0 = ld_frag16(&Ps[0][0], 64 + 8, w * 16, 0);
    v16h pa1 = ld_frag16(&Ps[0][0], 64 + 8, w * 16, 32);
#pragma unroll
    for (int ns = 0; ns < 4; ++ns) {
      o[ns] = wmma16(pa0, ld_frag16(&Vt[0][0], 64 + 8, ns * 16, 0), o[ns]);
      o[ns] = wmma16(pa1, ld_frag16(&Vt[0][0], 64 + 8, ns * 16, 32), o[ns]);
    }
  }

  _Float16* ab = aout + (size_t)(b * S_ + q0) * Dm + hh * DH;
#pragma unroll
  for (int ns = 0; ns < 4; ++ns) {
    int col = ns * 16 + (lane & 15);
#pragma unroll
    for (int r = 0; r < 8; ++r) {
      int row = w * 16 + r + 8 * grp;
      ab[(size_t)row * Dm + col] = (_Float16)(o[ns][r] / l_run[r]);
    }
  }
}

// ---------------------------------------------------------------------------
extern "C" void kernel_launch(void* const* d_in, const int* in_sizes, int n_in,
                              void* d_out, int out_size, void* d_ws,
                              size_t ws_size, hipStream_t stream) {
  (void)in_sizes; (void)n_in; (void)out_size; (void)ws_size;
  const int* ids = (const int*)d_in[0];
  const float* wte = (const float*)d_in[1];
  const float* wpe = (const float*)d_in[2];
  const float* ln1_g = (const float*)d_in[3];
  const float* ln1_b = (const float*)d_in[4];
  const float* w_qkv = (const float*)d_in[5];
  const float* b_qkv = (const float*)d_in[6];
  const float* w_ap = (const float*)d_in[7];
  const float* b_ap = (const float*)d_in[8];
  const float* ln2_g = (const float*)d_in[9];
  const float* ln2_b = (const float*)d_in[10];
  const float* w_fc = (const float*)d_in[11];
  const float* b_fc = (const float*)d_in[12];
  const float* w_mp = (const float*)d_in[13];
  const float* b_mp = (const float*)d_in[14];
  const float* lnf_g = (const float*)d_in[15];
  const float* lnf_b = (const float*)d_in[16];

  const size_t T = (size_t)B_ * S_;  // 2048 token rows
  float* h = (float*)d_ws;                    // f32 residual stream
  _Float16* x16 = (_Float16*)(h + T * Dm);    // ln output
  _Float16* qkv16 = x16 + T * Dm;
  _Float16* att16 = qkv16 + T * 3 * Dm;
  _Float16* ff16 = att16 + T * Dm;
  _Float16* wt_qkv = ff16 + T * DFF;          // per-layer f16 [N][K] weights
  _Float16* wt_ap = wt_qkv + (size_t)Dm * 3 * Dm;
  _Float16* wt_fc = wt_ap + (size_t)Dm * Dm;
  _Float16* wt_mp = wt_fc + (size_t)Dm * DFF;  // total ws ~= 68 MB

  const int M = (int)T;
  embed_kernel<<<(M * Dm) / 256, 256, 0, stream>>>(ids, wte, wpe, h);

  for (int l = 0; l < NL; ++l) {
    // per-layer weight convert+transpose (stream-ordered after prior users)
    wcvt_kernel<<<dim3(3 * Dm / 64, Dm / 64), 256, 0, stream>>>(
        w_qkv + (size_t)l * Dm * 3 * Dm, wt_qkv, Dm, 3 * Dm);
    wcvt_kernel<<<dim3(Dm / 64, Dm / 64), 256, 0, stream>>>(
        w_ap + (size_t)l * Dm * Dm, wt_ap, Dm, Dm);
    wcvt_kernel<<<dim3(DFF / 64, Dm / 64), 256, 0, stream>>>(
        w_fc + (size_t)l * Dm * DFF, wt_fc, Dm, DFF);
    wcvt_kernel<<<dim3(Dm / 64, DFF / 64), 256, 0, stream>>>(
        w_mp + (size_t)l * DFF * Dm, wt_mp, DFF, Dm);

    ln_kernel<_Float16><<<M, 256, 0, stream>>>(h, ln1_g + (size_t)l * Dm,
                                               ln1_b + (size_t)l * Dm, x16);
    gemm16_kernel<false, false, true>
        <<<dim3(3 * Dm / 128, M / 128), 256, 0, stream>>>(
            x16, wt_qkv, b_qkv + (size_t)l * 3 * Dm, nullptr, qkv16, M,
            3 * Dm, Dm);
    attn_kernel<<<dim3(S_ / 64, B_ * NH), 128, 0, stream>>>(qkv16, att16);
    gemm16_kernel<false, true, false>
        <<<dim3(Dm / 128, M / 128), 256, 0, stream>>>(
            att16, wt_ap, b_ap + (size_t)l * Dm, h, h, M, Dm, Dm);
    ln_kernel<_Float16><<<M, 256, 0, stream>>>(h, ln2_g + (size_t)l * Dm,
                                               ln2_b + (size_t)l * Dm, x16);
    gemm16_kernel<true, false, true>
        <<<dim3(DFF / 128, M / 128), 256, 0, stream>>>(
            x16, wt_fc, b_fc + (size_t)l * DFF, nullptr, ff16, M, DFF, Dm);
    gemm16_kernel<false, true, false>
        <<<dim3(Dm / 128, M / 128), 256, 0, stream>>>(
            ff16, wt_mp, b_mp + (size_t)l * Dm, h, h, M, Dm, DFF);
  }

  ln_kernel<float><<<M, 256, 0, stream>>>(h, lnf_g, lnf_b, (float*)d_out);
}
